// MLA_20787641712851
// MI455X (gfx1250) — matrix operand order repaired and checked by reading, not compiled, not run on hardware
//
#include <hip/hip_runtime.h>
#include <hip/hip_fp16.h>

typedef __attribute__((ext_vector_type(16))) _Float16 v16h;
typedef __attribute__((ext_vector_type(8)))  float    v8f;

#define B_   2
#define T_   2048
#define D_   2048
#define H_   16
#define HD_  192
#define RD_  64
#define CD_  128
#define QR_  768
#define DL_  512
#define ROWS (B_ * T_)   // 4096

#define WMMA_F16(Afr, Bfr, Cfr) \
  __builtin_amdgcn_wmma_f32_16x16x32_f16(false, (Afr), false, (Bfr), (short)0, (Cfr), false, false)

// A-matrix (16x32 f16) element->K mapping for wave32 WMMA fragments.
// v16h element j: K = 16*half + j (contiguous, matches B-fragment fills).
__device__ __forceinline__ int a_kidx(int j, int half) {
  int i = j >> 1;
  (void)i;
  return (half << 4) + j;
}

// Async global->LDS copy (16B per lane), ASYNCcnt-tracked. GVS addressing:
// 32-bit per-lane byte offset VGPR + uniform 64-bit SGPR base.
__device__ __forceinline__ void async_ld_b128(unsigned lds_addr, unsigned goff, const void* base) {
  asm volatile("global_load_async_to_lds_b128 %0, %1, %2"
               :: "v"(lds_addr), "v"(goff), "s"(base) : "memory");
}
template <int N>
__device__ __forceinline__ void wait_async() {
  asm volatile("s_wait_asynccnt %0" :: "n"(N) : "memory");
}
// Low 32 bits of a flat LDS pointer are the LDS byte offset (aperture layout).
__device__ __forceinline__ unsigned lds_off(const void* p) {
  return (unsigned)(size_t)p;
}

// ---------------------------------------------------------------------------
// fp32 -> fp16 elementwise conversion (one-time pre-pass; bandwidth trivial)
// ---------------------------------------------------------------------------
__global__ __launch_bounds__(256)
void f32_to_f16(const float* __restrict__ s, _Float16* __restrict__ d, int n) {
  int i = blockIdx.x * blockDim.x + threadIdx.x;
  int stride = gridDim.x * blockDim.x;
  for (; i < n; i += stride) d[i] = (_Float16)s[i];
}

// ---------------------------------------------------------------------------
// Tiled GEMM, C = A(MxK) * B(KxN), all-f16 operands, fp32 WMMA accumulate.
// Block 256 threads (8 waves), tile 128x128, BK=32, double-buffered LDS
// filled with async global->LDS B128 copies. Wave (w>>1, w&1) -> 32x64.
// ---------------------------------------------------------------------------
template <bool OUT_HALF>
__global__ __launch_bounds__(256)
void gemm_wmma_f16(const _Float16* __restrict__ A, const _Float16* __restrict__ Bm,
                   void* __restrict__ Cv, int M, int N, int K) {
  __shared__ __align__(16) _Float16 sA[2][128][40];   // 32 payload + 8 pad per row
  __shared__ __align__(16) _Float16 sB[2][32][136];   // 128 payload + 8 pad per row

  const int tid  = threadIdx.x;
  const int wave = tid >> 5;
  const int lane = tid & 31;
  const int half = lane >> 4;
  const int ln   = lane & 15;

  const int bm = blockIdx.y * 128;
  const int bn = blockIdx.x * 128;
  const int wm = (wave >> 1) * 32;
  const int wn = (wave & 1) * 64;

  v8f acc[2][4];
#pragma unroll
  for (int i = 0; i < 2; ++i)
#pragma unroll
    for (int j = 0; j < 4; ++j)
#pragma unroll
      for (int r = 0; r < 8; ++r) acc[i][j][r] = 0.0f;

  auto issue = [&](int buf, int k0) {
    // A tile: 128 rows x 32 f16 = 512 16B-chunks (4 per row)
#pragma unroll
    for (int t = 0; t < 2; ++t) {
      int e = tid + t * 256;
      int r = e >> 2, cc = e & 3;
      async_ld_b128(lds_off(&sA[buf][r][cc * 8]),
                    (unsigned)(((bm + r) * K + k0) * 2 + cc * 16), A);
    }
    // B tile: 32 rows x 128 f16 = 512 16B-chunks (16 per row)
#pragma unroll
    for (int t = 0; t < 2; ++t) {
      int e = tid + t * 256;
      int r = e >> 4, cc = e & 15;
      async_ld_b128(lds_off(&sB[buf][r][cc * 8]),
                    (unsigned)(((k0 + r) * N + bn) * 2 + cc * 16), Bm);
    }
  };

  const int nk = K / 32;
  issue(0, 0);
  int cur = 0;
  for (int kk = 0; kk < nk; ++kk) {
    __syncthreads();                       // prior compute done before refilling other buffer
    if (kk + 1 < nk) { issue(cur ^ 1, (kk + 1) * 32); wait_async<4>(); }
    else             { wait_async<0>(); }
    __syncthreads();                       // tile kk resident in LDS for all waves

    v16h afr[2];
#pragma unroll
    for (int s = 0; s < 2; ++s)
#pragma unroll
      for (int j = 0; j < 16; ++j)
        afr[s][j] = sA[cur][wm + s * 16 + ln][a_kidx(j, half)];
#pragma unroll
    for (int sn = 0; sn < 4; ++sn) {
      v16h bfr;
#pragma unroll
      for (int j = 0; j < 16; ++j)
        bfr[j] = sB[cur][(half << 4) + j][wn + sn * 16 + ln];
#pragma unroll
      for (int si = 0; si < 2; ++si)
        acc[si][sn] = WMMA_F16(afr[si], bfr, acc[si][sn]);
    }
    cur ^= 1;
  }

#pragma unroll
  for (int si = 0; si < 2; ++si)
#pragma unroll
    for (int sn = 0; sn < 4; ++sn)
#pragma unroll
      for (int r = 0; r < 8; ++r) {
        int row = bm + wm + 16 * si + 8 * half + r;
        int col = bn + wn + 16 * sn + ln;
        if constexpr (OUT_HALF)
          ((_Float16*)Cv)[(size_t)row * N + col] = (_Float16)acc[si][sn][r];
        else
          ((float*)Cv)[(size_t)row * N + col] = acc[si][sn][r];
      }
}

// ---------------------------------------------------------------------------
// Fused RoPE + head-interleave pack: build qfull/kfull (ROWS, H, 192) f16 from
// content (ROWS, H*128) and rope (ROWS, H*64) parts.
// ---------------------------------------------------------------------------
__global__ __launch_bounds__(256)
void rope_pack(const _Float16* __restrict__ qc, const _Float16* __restrict__ qp,
               const _Float16* __restrict__ kc, const _Float16* __restrict__ kp,
               _Float16* __restrict__ qfull, _Float16* __restrict__ kfull) {
  int idx = blockIdx.x * blockDim.x + threadIdx.x;   // ROWS*H*192
  if (idx >= ROWS * H_ * HD_) return;
  int d   = idx % HD_;
  int h   = (idx / HD_) % H_;
  int row = idx / (HD_ * H_);
  int t   = row & (T_ - 1);
  size_t o = ((size_t)row * H_ + h) * HD_ + d;
  if (d < CD_) {
    qfull[o] = qc[(size_t)row * (H_ * CD_) + h * CD_ + d];
    kfull[o] = kc[(size_t)row * (H_ * CD_) + h * CD_ + d];
  } else {
    int dr = d - CD_;          // 0..63
    int dd = dr & 31;          // rotary pair index
    bool hi = dr >= 32;
    float inv = __powf(10000.0f, -(float)dd / 32.0f);
    float ang = (float)t * inv;
    float c = __cosf(ang), s = __sinf(ang);
    size_t base = (size_t)row * (H_ * RD_) + h * RD_ + dd;
    float x1 = (float)qp[base], x2 = (float)qp[base + 32];
    qfull[o] = (_Float16)(hi ? (x1 * s + x2 * c) : (x1 * c - x2 * s));
    x1 = (float)kp[base]; x2 = (float)kp[base + 32];
    kfull[o] = (_Float16)(hi ? (x1 * s + x2 * c) : (x1 * c - x2 * s));
  }
}

// ---------------------------------------------------------------------------
// Causal flash attention. 256 threads = 8 waves; each wave owns 16 query rows
// (block covers 128 queries of one (b,h)). Key tiles of 32, online softmax.
// K/V tiles staged via async global->LDS copies.
// ---------------------------------------------------------------------------
__global__ __launch_bounds__(256)
void mla_attn(const _Float16* __restrict__ qfull, const _Float16* __restrict__ kfull,
              const _Float16* __restrict__ vb, _Float16* __restrict__ ao) {
  __shared__ __align__(16) _Float16 sK[32][200];   // 192 payload + 8 pad
  __shared__ __align__(16) _Float16 sV[32][200];
  __shared__ __align__(16) _Float16 sP[8][16][34];

  const int tid  = threadIdx.x;
  const int wave = tid >> 5;
  const int lane = tid & 31;
  const int half = lane >> 4;
  const int ln   = lane & 15;

  const int qtiles = T_ / 128;
  int bid = blockIdx.x;
  const int qt = bid % qtiles; bid /= qtiles;
  const int h  = bid % H_;
  const int b  = bid / H_;
  const int qbase = qt * 128;

  // Q fragments from qfull (contiguous per (row,h)): 6 K-chunks of 32 over 192
  const size_t rowQ = ((size_t)(b * T_ + qbase + wave * 16 + ln) * H_ + h) * HD_;
  v16h qf[6];
#pragma unroll
  for (int c = 0; c < 6; ++c)
#pragma unroll
    for (int j = 0; j < 16; ++j)
      qf[c][j] = qfull[rowQ + c * 32 + a_kidx(j, half)];

  v8f o[12];
#pragma unroll
  for (int t = 0; t < 12; ++t)
#pragma unroll
    for (int r = 0; r < 8; ++r) o[t][r] = 0.0f;
  float mst[8], lst[8];
#pragma unroll
  for (int r = 0; r < 8; ++r) { mst[r] = -3.0e38f; lst[r] = 0.0f; }

  const float scale = 0.07216878364870323f;  // 1/sqrt(192)
  const int ktiles = (qbase + 128) / 32;

  for (int kt = 0; kt < ktiles; ++kt) {
    const int kb = kt * 32;
    // Async stage K/V: 32 rows x 192 f16 = 768 16B-chunks each (24 per row)
#pragma unroll
    for (int t = 0; t < 3; ++t) {
      int e = tid + t * 256;
      int r = e / 24, cc = e % 24;
      unsigned go = (unsigned)((((b * T_ + kb + r) * H_ + h) * HD_) * 2 + cc * 16);
      async_ld_b128(lds_off(&sK[r][cc * 8]), go, kfull);
      async_ld_b128(lds_off(&sV[r][cc * 8]), go, vb);
    }
    wait_async<0>();
    __syncthreads();

    // S = Q * K^T  (two 16-key subtiles)
    v8f s0, s1;
#pragma unroll
    for (int r = 0; r < 8; ++r) { s0[r] = 0.0f; s1[r] = 0.0f; }
#pragma unroll
    for (int c = 0; c < 6; ++c) {
      v16h b0, b1;
#pragma unroll
      for (int j = 0; j < 16; ++j) {
        int d = c * 32 + 16 * half + j;
        b0[j] = sK[ln][d];
        b1[j] = sK[16 + ln][d];
      }
      s0 = WMMA_F16(qf[c], b0, s0);
      s1 = WMMA_F16(qf[c], b1, s1);
    }

    // Scale + causal mask
    const int qrow0 = qbase + wave * 16 + 8 * half;
#pragma unroll
    for (int r = 0; r < 8; ++r) {
      int qg = qrow0 + r;
      float v0 = s0[r] * scale;
      float v1 = s1[r] * scale;
      if (kb + ln > qg)      v0 = -3.0e38f;
      if (kb + 16 + ln > qg) v1 = -3.0e38f;
      s0[r] = v0; s1[r] = v1;
    }

    // Online softmax (row reductions across the 16 lanes of each half)
#pragma unroll
    for (int r = 0; r < 8; ++r) {
      float t = fmaxf(s0[r], s1[r]);
      t = fmaxf(t, __shfl_xor(t, 1, 32));
      t = fmaxf(t, __shfl_xor(t, 2, 32));
      t = fmaxf(t, __shfl_xor(t, 4, 32));
      t = fmaxf(t, __shfl_xor(t, 8, 32));
      float mnew = fmaxf(mst[r], t);
      float p0 = __expf(s0[r] - mnew);
      float p1 = __expf(s1[r] - mnew);
      s0[r] = p0; s1[r] = p1;
      float rs = p0 + p1;
      rs += __shfl_xor(rs, 1, 32);
      rs += __shfl_xor(rs, 2, 32);
      rs += __shfl_xor(rs, 4, 32);
      rs += __shfl_xor(rs, 8, 32);
      float f = __expf(mst[r] - mnew);
      lst[r] = lst[r] * f + rs;
      mst[r] = mnew;
#pragma unroll
      for (int t2 = 0; t2 < 12; ++t2) o[t2][r] *= f;
    }

    // Re-layout P: D-fragment -> per-wave LDS -> A-fragment
#pragma unroll
    for (int r = 0; r < 8; ++r) {
      sP[wave][8 * half + r][ln]      = (_Float16)s0[r];
      sP[wave][8 * half + r][16 + ln] = (_Float16)s1[r];
    }
    __builtin_amdgcn_wave_barrier();
    v16h pf;
#pragma unroll
    for (int j = 0; j < 16; ++j) pf[j] = sP[wave][ln][a_kidx(j, half)];

    // O += P * V (12 column blocks of 16)
#pragma unroll
    for (int nb = 0; nb < 12; ++nb) {
      v16h vf;
#pragma unroll
      for (int j = 0; j < 16; ++j) vf[j] = sV[16 * half + j][nb * 16 + ln];
      o[nb] = WMMA_F16(pf, vf, o[nb]);
    }
    __syncthreads();   // all waves done reading sK/sV before next async fill
  }

  // Normalize and store attn output (ROWS, H*HD) f16
#pragma unroll
  for (int nb = 0; nb < 12; ++nb)
#pragma unroll
    for (int r = 0; r < 8; ++r) {
      int qg  = qbase + wave * 16 + 8 * half + r;
      int col = h * HD_ + nb * 16 + ln;
      ao[(size_t)(b * T_ + qg) * (H_ * HD_) + col] = (_Float16)(o[nb][r] / lst[r]);
    }
}

// ---------------------------------------------------------------------------
// Host-side orchestration
// ---------------------------------------------------------------------------
static inline size_t align256(size_t x) { return (x + 255) & ~(size_t)255; }

extern "C" void kernel_launch(void* const* d_in, const int* in_sizes, int n_in,
                              void* d_out, int out_size, void* d_ws, size_t ws_size,
                              hipStream_t stream) {
  (void)in_sizes; (void)n_in; (void)out_size; (void)ws_size;
  const float* x        = (const float*)d_in[0];
  const float* wq_down  = (const float*)d_in[1];
  const float* wq_up    = (const float*)d_in[2];
  const float* wq_rope  = (const float*)d_in[3];
  const float* wkv_down = (const float*)d_in[4];
  const float* wk_up    = (const float*)d_in[5];
  const float* wv_up    = (const float*)d_in[6];
  const float* wk_rope  = (const float*)d_in[7];
  const float* wo       = (const float*)d_in[8];
  float* out = (float*)d_out;

  char* p = (char*)d_ws;
  auto carve = [&](size_t elems) { _Float16* q = (_Float16*)p; p += align256(elems * 2); return q; };
  _Float16* xh     = carve((size_t)ROWS * D_);
  _Float16* wdh    = carve((size_t)D_ * QR_);
  _Float16* wuh    = carve((size_t)QR_ * H_ * CD_);
  _Float16* wrh    = carve((size_t)D_ * H_ * RD_);
  _Float16* wkvh   = carve((size_t)D_ * DL_);
  _Float16* wkuh   = carve((size_t)DL_ * H_ * CD_);
  _Float16* wvuh   = carve((size_t)DL_ * H_ * HD_);
  _Float16* wkrh   = carve((size_t)D_ * H_ * RD_);
  _Float16* woh    = carve((size_t)H_ * HD_ * D_);
  _Float16* tmpq   = carve((size_t)ROWS * QR_);
  _Float16* qc     = carve((size_t)ROWS * H_ * CD_);
  _Float16* qp     = carve((size_t)ROWS * H_ * RD_);
  _Float16* latent = carve((size_t)ROWS * DL_);
  _Float16* kc     = carve((size_t)ROWS * H_ * CD_);
  _Float16* kp     = carve((size_t)ROWS * H_ * RD_);
  _Float16* vbuf   = carve((size_t)ROWS * H_ * HD_);
  _Float16* qfull  = carve((size_t)ROWS * H_ * HD_);
  _Float16* kfull  = carve((size_t)ROWS * H_ * HD_);
  _Float16* ao     = carve((size_t)ROWS * H_ * HD_);

  auto cvt = [&](const float* s, _Float16* d, size_t n) {
    int blocks = (int)((n + 255) / 256); if (blocks > 4096) blocks = 4096;
    f32_to_f16<<<blocks, 256, 0, stream>>>(s, d, (int)n);
  };
  cvt(x, xh, (size_t)ROWS * D_);
  cvt(wq_down, wdh, (size_t)D_ * QR_);
  cvt(wq_up, wuh, (size_t)QR_ * H_ * CD_);
  cvt(wq_rope, wrh, (size_t)D_ * H_ * RD_);
  cvt(wkv_down, wkvh, (size_t)D_ * DL_);
  cvt(wk_up, wkuh, (size_t)DL_ * H_ * CD_);
  cvt(wv_up, wvuh, (size_t)DL_ * H_ * HD_);
  cvt(wk_rope, wkrh, (size_t)D_ * H_ * RD_);
  cvt(wo, woh, (size_t)H_ * HD_ * D_);

  dim3 blk(256);
  // 1) tmpq = xh @ wq_down            (4096x2048)*(2048x768)
  gemm_wmma_f16<true><<<dim3(QR_ / 128, ROWS / 128), blk, 0, stream>>>(xh, wdh, tmpq, ROWS, QR_, D_);
  // 2) qc = tmpq @ wq_up              (4096x768)*(768x2048)
  gemm_wmma_f16<true><<<dim3((H_ * CD_) / 128, ROWS / 128), blk, 0, stream>>>(tmpq, wuh, qc, ROWS, H_ * CD_, QR_);
  // 3) qp = xh @ wq_rope              (4096x2048)*(2048x1024)
  gemm_wmma_f16<true><<<dim3((H_ * RD_) / 128, ROWS / 128), blk, 0, stream>>>(xh, wrh, qp, ROWS, H_ * RD_, D_);
  // 4) latent = xh @ wkv_down         (4096x2048)*(2048x512)
  gemm_wmma_f16<true><<<dim3(DL_ / 128, ROWS / 128), blk, 0, stream>>>(xh, wkvh, latent, ROWS, DL_, D_);
  // 5) kc = latent @ wk_up            (4096x512)*(512x2048)
  gemm_wmma_f16<true><<<dim3((H_ * CD_) / 128, ROWS / 128), blk, 0, stream>>>(latent, wkuh, kc, ROWS, H_ * CD_, DL_);
  // 6) kp = xh @ wk_rope              (4096x2048)*(2048x1024)
  gemm_wmma_f16<true><<<dim3((H_ * RD_) / 128, ROWS / 128), blk, 0, stream>>>(xh, wkrh, kp, ROWS, H_ * RD_, D_);
  // 7) v = latent @ wv_up             (4096x512)*(512x3072)
  gemm_wmma_f16<true><<<dim3((H_ * HD_) / 128, ROWS / 128), blk, 0, stream>>>(latent, wvuh, vbuf, ROWS, H_ * HD_, DL_);

  // 8) RoPE + pack q/k into (ROWS, H, 192) f16
  {
    int total = ROWS * H_ * HD_;
    rope_pack<<<(total + 255) / 256, 256, 0, stream>>>(qc, qp, kc, kp, qfull, kfull);
  }

  // 9) Causal flash attention -> ao
  mla_attn<<<B_ * H_ * (T_ / 128), 256, 0, stream>>>(qfull, kfull, vbuf, ao);

  // 10) out = ao @ wo                 (4096x3072)*(3072x2048) -> fp32
  gemm_wmma_f16<false><<<dim3(D_ / 128, ROWS / 128), blk, 0, stream>>>(ao, woh, out, ROWS, D_, H_ * HD_);
}
